// PointTransformer_15126874816983
// MI455X (gfx1250) — matrix operand order repaired and checked by reading, not compiled, not run on hardware
//
#include <hip/hip_runtime.h>
#include <hip/hip_bf16.h>

typedef __attribute__((ext_vector_type(16))) _Float16 v16h;
typedef __attribute__((ext_vector_type(8)))  _Float16 v8h;
typedef __attribute__((ext_vector_type(8)))  float    v8f;
typedef __attribute__((ext_vector_type(4)))  float    v4f;
typedef __attribute__((ext_vector_type(4)))  int      v4i;

#define DEVFN static __device__ __forceinline__

// Detect the gfx1250 async global->LDS DMA builtins (arity/availability varies
// across ROCm toolchains); fall back to plain loads when absent.
#define HAVE_ASYNC_LDS 0
#ifdef __has_builtin
#if __has_builtin(__builtin_amdgcn_global_load_async_to_lds_b128)
#undef HAVE_ASYNC_LDS
#define HAVE_ASYNC_LDS 1
#endif
#endif

#if HAVE_ASYNC_LDS
typedef __attribute__((address_space(3))) v4i v4i_lds;
#endif

constexpr int B_  = 2;
constexpr int N_  = 2048;
constexpr int DP  = 32;     // d_points
constexpr int DM  = 512;    // d_model
constexpr int KNN = 16;
constexpr int BN  = B_ * N_;        // 4096 rows
constexpr int RKN = BN * KNN;       // 65536 neighbor rows

// ---------------- WMMA fragment helpers (layouts per CDNA5 ISA 7.12.2) -----

DEVFN v8f wmma_f16(v16h a, v16h b, v8f c) {
  return __builtin_amdgcn_wmma_f32_16x16x32_f16(
      /*neg_a=*/false, a, /*neg_b=*/false, b,
      /*c_mod=*/(short)0, c, /*reuse_a=*/false, /*reuse_b=*/false);
}

// A fragment (16x32) from row-major f32 source; Abase points at tile row 0.
// Two contiguous float4 pairs per lane -> b128 loads.
DEVFN v16h a_frag_f32(const float* __restrict__ Abase, int lda, int k0, int lane) {
  int h = lane >> 4, m = lane & 15;
  const float* row = Abase + (size_t)m * lda + k0;
  v4f a0 = *(const v4f*)(row + h * 8);
  v4f a1 = *(const v4f*)(row + h * 8 + 4);
  v4f a2 = *(const v4f*)(row + 16 + h * 8);
  v4f a3 = *(const v4f*)(row + 16 + h * 8 + 4);
  v16h a;
#pragma unroll
  for (int i = 0; i < 4; ++i) {
    a[i]      = (_Float16)a0[i];
    a[4 + i]  = (_Float16)a1[i];
    a[8 + i]  = (_Float16)a2[i];
    a[12 + i] = (_Float16)a3[i];
  }
  return a;
}

// A fragment (16x32) from row-major f16 source (global or LDS): 2 x b128.
DEVFN v16h a_frag_h(const _Float16* __restrict__ Abase, int lda, int k0, int lane) {
  int h = lane >> 4, m = lane & 15;
  const _Float16* row = Abase + (size_t)m * lda + k0;
  v8h lo = *(const v8h*)(row + h * 8);
  v8h hi = *(const v8h*)(row + 16 + h * 8);
  v16h a;
#pragma unroll
  for (int i = 0; i < 8; ++i) { a[i] = lo[i]; a[8 + i] = hi[i]; }
  return a;
}

// B fragment (32x16) from TRANSPOSED f16 weights Wt[N][K]: 2 x b128 per lane.
DEVFN v16h b_frag_t(const _Float16* __restrict__ Wt, int ldk, int k0, int n0, int lane) {
  int h = lane >> 4, n = lane & 15;
  const _Float16* col = Wt + (size_t)(n0 + n) * ldk + k0;
  v8h lo = *(const v8h*)(col + h * 8);
  v8h hi = *(const v8h*)(col + 16 + h * 8);
  v16h b;
#pragma unroll
  for (int i = 0; i < 8; ++i) { b[i] = lo[i]; b[8 + i] = hi[i]; }
  return b;
}

// ---------------- Kernel W: transpose + f32->f16 convert weights -----------

__global__ __launch_bounds__(256)
void wT_kernel(const float* __restrict__ W, _Float16* __restrict__ Wt, int K, int Nn) {
  int t = blockIdx.x * 256 + threadIdx.x;
  if (t >= K * Nn) return;
  int k = t / Nn, n = t % Nn;                 // coalesced read of W[k][n]
  Wt[(size_t)n * K + k] = (_Float16)W[t];
}

// ---------------- Kernel 0: KNN top-16 by squared distance -----------------

__global__ __launch_bounds__(32)
void knn_kernel(const float* __restrict__ xyz, int* __restrict__ knn) {
  __shared__ float distS[N_];
  const int row  = blockIdx.x;          // 0..BN-1
  const int bat  = row / N_;
  const int lane = threadIdx.x;
  const float* base = xyz + (size_t)bat * N_ * 3;
  const float qx = xyz[(size_t)row * 3 + 0];
  const float qy = xyz[(size_t)row * 3 + 1];
  const float qz = xyz[(size_t)row * 3 + 2];
  for (int j = lane; j < N_; j += 32) {
    float dx = qx - base[j * 3 + 0];
    float dy = qy - base[j * 3 + 1];
    float dz = qz - base[j * 3 + 2];
    distS[j] = dx * dx + dy * dy + dz * dz;
  }
  __syncthreads();
  for (int p = 0; p < KNN; ++p) {
    float best = 3.4e38f; int bidx = N_;
    for (int j = lane; j < N_; j += 32) {
      float d = distS[j];
      if (d < best) { best = d; bidx = j; }
    }
#pragma unroll
    for (int off = 16; off > 0; off >>= 1) {
      float ov = __shfl_xor(best, off, 32);
      int   oi = __shfl_xor(bidx, off, 32);
      if (ov < best || (ov == best && oi < bidx)) { best = ov; bidx = oi; }
    }
    if (lane == 0) {
      knn[(size_t)row * KNN + p] = bidx;
      distS[bidx] = 3.4e38f;
    }
    __syncthreads();
  }
}

// ---------------- Kernel 1: x = points @ fc1_w + fc1_b  (K=32, one WMMA) ---

__global__ __launch_bounds__(256)
void fc1_kernel(const float* __restrict__ pts, const _Float16* __restrict__ wt,
                const float* __restrict__ bias, _Float16* __restrict__ xh) {
  const int wid  = (blockIdx.x * 256 + threadIdx.x) >> 5;   // wave id
  const int lane = threadIdx.x & 31;
  const int tm = wid >> 5, tn = wid & 31;                   // 256 x 32 tiles
  const int r0 = tm * 16, c0 = tn * 16;
  v16h a  = a_frag_f32(pts + (size_t)r0 * DP, DP, 0, lane);
  v16h bf = b_frag_t(wt, DP, 0, c0, lane);                  // Wt[512][32]
  v8f acc = {};
  acc = wmma_f16(a, bf, acc);
  const int h = lane >> 4, n = lane & 15;
  const float bb = bias[c0 + n];
#pragma unroll
  for (int i = 0; i < 8; ++i)
    xh[(size_t)(r0 + h * 8 + i) * DM + c0 + n] = (_Float16)(acc[i] + bb);
}

// ---------------- Kernel 2: out = xh(f16) @ W (512x512), no bias -----------
// One wave computes a 16x64 strip (A fragment reused across 4 tiles).

__global__ __launch_bounds__(256)
void lin512_kernel(const _Float16* __restrict__ xh, const _Float16* __restrict__ wt,
                   float* __restrict__ out) {
  const int wid  = (blockIdx.x * 256 + threadIdx.x) >> 5;   // 2048 waves
  const int lane = threadIdx.x & 31;
  const int tm = wid >> 3, cg = wid & 7;
  const int r0 = tm * 16, cbase = cg * 64;
  v8f acc[4] = {};
  for (int k0 = 0; k0 < DM; k0 += 32) {
    v16h a = a_frag_h(xh + (size_t)r0 * DM, DM, k0, lane);
#pragma unroll
    for (int j = 0; j < 4; ++j) {
      v16h bf = b_frag_t(wt, DM, k0, cbase + j * 16, lane);
      acc[j] = wmma_f16(a, bf, acc[j]);
    }
  }
  const int h = lane >> 4, n = lane & 15;
#pragma unroll
  for (int j = 0; j < 4; ++j)
#pragma unroll
    for (int i = 0; i < 8; ++i)
      out[(size_t)(r0 + h * 8 + i) * DM + cbase + j * 16 + n] = acc[j][i];
}

// ---------------- Kernel 3: pos_enc = relu(delta@w1+b1)@w2 + b2  (f16 out) -

__global__ __launch_bounds__(256)
void pos_kernel(const float* __restrict__ xyz, const int* __restrict__ knn,
                const float* __restrict__ w1, const float* __restrict__ b1,
                const _Float16* __restrict__ w2t, const float* __restrict__ b2,
                _Float16* __restrict__ pos) {
  __shared__ __align__(16) _Float16 hid[64 * DM];     // 64 KB
  __shared__ float dS[64][4];
  const int t = threadIdx.x;
  const int rowblk = blockIdx.x * 64;
  if (t < 64) {
    int r = rowblk + t;
    int bn = r / KNN, bat = bn / N_;
    int idx = knn[r];
    const float* pq = xyz + (size_t)bn * 3;
    const float* pk = xyz + ((size_t)bat * N_ + idx) * 3;
    dS[t][0] = pq[0] - pk[0];
    dS[t][1] = pq[1] - pk[1];
    dS[t][2] = pq[2] - pk[2];
  }
  __syncthreads();
  // hidden = relu(delta @ w1 + b1): tiny K=3 layer in VALU
  for (int i = t; i < 64 * DM; i += 256) {
    int rr = i >> 9, c = i & (DM - 1);
    float s = b1[c] + dS[rr][0] * w1[c] + dS[rr][1] * w1[DM + c] + dS[rr][2] * w1[2 * DM + c];
    hid[i] = (_Float16)(s > 0.f ? s : 0.f);
  }
  __syncthreads();
  // 8 waves: (row strip 0..3) x (column half 0..1)
  const int wv = t >> 5, lane = t & 31;
  const int r0 = (wv & 3) * 16;
  const int chalf = (wv >> 2) * 256;
  const int h = lane >> 4, n = lane & 15;
  for (int nc = 0; nc < 256; nc += 64) {
    v8f acc[4] = {};
    for (int k0 = 0; k0 < DM; k0 += 32) {
      v16h a = a_frag_h(hid + (size_t)r0 * DM, DM, k0, lane);
#pragma unroll
      for (int j = 0; j < 4; ++j) {
        v16h bf = b_frag_t(w2t, DM, k0, chalf + nc + j * 16, lane);
        acc[j] = wmma_f16(a, bf, acc[j]);
      }
    }
#pragma unroll
    for (int j = 0; j < 4; ++j) {
      int c0 = chalf + nc + j * 16;
      float bb = b2[c0 + n];
#pragma unroll
      for (int i = 0; i < 8; ++i)
        pos[(size_t)(rowblk + r0 + h * 8 + i) * DM + c0 + n] = (_Float16)(acc[j][i] + bb);
    }
  }
}

// ---------------- Kernel 4: gamma MLP on (q - k_gather + pos_enc) ----------

__global__ __launch_bounds__(256)
void gamma_kernel(const float* __restrict__ q, const float* __restrict__ kf,
                  const _Float16* __restrict__ pos, const int* __restrict__ knn,
                  const _Float16* __restrict__ w1t, const float* __restrict__ b1,
                  const _Float16* __restrict__ w2t, const float* __restrict__ b2,
                  float* __restrict__ aout) {
  __shared__ __align__(16) _Float16 ain[64 * DM];     // 64 KB
  __shared__ __align__(16) _Float16 hid[64 * DM];     // 64 KB
  const int t = threadIdx.x;
  const int rowblk = blockIdx.x * 64;

#if HAVE_ASYNC_LDS
  // Stage the contiguous 64 KB pos tile via the async DMA path, overlapping
  // with the q/k gather below.
  __shared__ __align__(16) _Float16 posS[64 * DM];    // 64 KB
  {
    const _Float16* src = pos + (size_t)rowblk * DM;
    for (int i = t; i < (64 * DM) / 8; i += 256) {
      __builtin_amdgcn_global_load_async_to_lds_b128(
          (v4i*)(src + (size_t)i * 8), (v4i_lds*)(posS + (size_t)i * 8), 0, 0);
    }
  }
#endif

  // Build a_in = q - k_gather + pos  (f16 tile in LDS)
  for (int i = t; i < 64 * DM; i += 256) {
    int rr = i >> 9, c = i & (DM - 1);
    int r = rowblk + rr;
    int bn = r / KNN, bat = bn / N_;
    int idx = knn[r];
    float qv = q[(size_t)bn * DM + c];
    float kv = kf[((size_t)bat * N_ + idx) * DM + c];
#if HAVE_ASYNC_LDS
    if (i == t) {   // wait once per thread, after gathers were issued
#if __has_builtin(__builtin_amdgcn_s_wait_asynccnt)
      __builtin_amdgcn_s_wait_asynccnt(0);
#else
      asm volatile("s_wait_asynccnt 0x0" ::: "memory");
#endif
    }
    float pv = (float)posS[i];
#else
    float pv = (float)pos[(size_t)r * DM + c];
#endif
    ain[i] = (_Float16)(qv - kv + pv);
  }
  __syncthreads();

  const int wv = t >> 5, lane = t & 31;
  const int r0 = (wv & 3) * 16;
  const int chalf = (wv >> 2) * 256;
  const int h = lane >> 4, n = lane & 15;

  // hidden = relu(ain @ w1 + b1)
  for (int nc = 0; nc < 256; nc += 64) {
    v8f acc[4] = {};
    for (int k0 = 0; k0 < DM; k0 += 32) {
      v16h a = a_frag_h(ain + (size_t)r0 * DM, DM, k0, lane);
#pragma unroll
      for (int j = 0; j < 4; ++j) {
        v16h bf = b_frag_t(w1t, DM, k0, chalf + nc + j * 16, lane);
        acc[j] = wmma_f16(a, bf, acc[j]);
      }
    }
#pragma unroll
    for (int j = 0; j < 4; ++j) {
      int c0 = chalf + nc + j * 16;
      float bb = b1[c0 + n];
#pragma unroll
      for (int i = 0; i < 8; ++i) {
        float s = acc[j][i] + bb;
        hid[(size_t)(r0 + h * 8 + i) * DM + c0 + n] = (_Float16)(s > 0.f ? s : 0.f);
      }
    }
  }
  __syncthreads();

  // a = hid @ w2 + b2  -> pre-softmax logits straight into d_out attn region
  for (int nc = 0; nc < 256; nc += 64) {
    v8f acc[4] = {};
    for (int k0 = 0; k0 < DM; k0 += 32) {
      v16h a = a_frag_h(hid + (size_t)r0 * DM, DM, k0, lane);
#pragma unroll
      for (int j = 0; j < 4; ++j) {
        v16h bf = b_frag_t(w2t, DM, k0, chalf + nc + j * 16, lane);
        acc[j] = wmma_f16(a, bf, acc[j]);
      }
    }
#pragma unroll
    for (int j = 0; j < 4; ++j) {
      int c0 = chalf + nc + j * 16;
      float bb = b2[c0 + n];
#pragma unroll
      for (int i = 0; i < 8; ++i)
        aout[(size_t)(rowblk + r0 + h * 8 + i) * DM + c0 + n] = acc[j][i] + bb;
    }
  }
}

// ---------------- Kernel 5: softmax over the 16 neighbors (per channel) ----

__global__ __launch_bounds__(256)
void softmax_kernel(float* __restrict__ attn) {
  int t = blockIdx.x * 256 + threadIdx.x;
  if (t >= BN * DM) return;
  int bn = t >> 9, c = t & (DM - 1);
  float* p = attn + (size_t)bn * KNN * DM + c;
  float vals[KNN];
  float m = -3.4e38f;
#pragma unroll
  for (int kk = 0; kk < KNN; ++kk) { vals[kk] = p[(size_t)kk * DM]; m = fmaxf(m, vals[kk]); }
  const float inv = 0.04419417382415922f;   // 1/sqrt(512)
  float s = 0.f;
#pragma unroll
  for (int kk = 0; kk < KNN; ++kk) { float e = __expf((vals[kk] - m) * inv); vals[kk] = e; s += e; }
  float rs = 1.f / s;
#pragma unroll
  for (int kk = 0; kk < KNN; ++kk) p[(size_t)kk * DM] = vals[kk] * rs;
}

// ---------------- Kernel 6: res = sum_k attn * (v_gather + pos_enc) --------

__global__ __launch_bounds__(256)
void resmid_kernel(const float* __restrict__ attn, const float* __restrict__ vf,
                   const _Float16* __restrict__ pos, const int* __restrict__ knn,
                   float* __restrict__ resmid) {
  int t = blockIdx.x * 256 + threadIdx.x;
  if (t >= BN * DM) return;
  int bn = t >> 9, c = t & (DM - 1);
  int bat = bn / N_;
  float acc = 0.f;
#pragma unroll
  for (int kk = 0; kk < KNN; ++kk) {
    int r = bn * KNN + kk;
    int idx = knn[r];
    acc += attn[(size_t)r * DM + c] *
           (vf[((size_t)bat * N_ + idx) * DM + c] + (float)pos[(size_t)r * DM + c]);
  }
  resmid[(size_t)bn * DM + c] = acc;
}

// ---------------- Kernel 7: out = resmid @ fc2_w + fc2_b + points ----------

__global__ __launch_bounds__(256)
void fc2_kernel(const float* __restrict__ resmid, const _Float16* __restrict__ wt,
                const float* __restrict__ bias, const float* __restrict__ pts,
                float* __restrict__ out) {
  const int wid  = (blockIdx.x * 256 + threadIdx.x) >> 5;   // 512 waves total
  const int lane = threadIdx.x & 31;
  const int tm = wid >> 1, tn = wid & 1;                    // 256 x 2 tiles
  const int r0 = tm * 16, c0 = tn * 16;
  v8f acc = {};
  for (int k0 = 0; k0 < DM; k0 += 32) {
    v16h a  = a_frag_f32(resmid + (size_t)r0 * DM, DM, k0, lane);
    v16h bf = b_frag_t(wt, DM, k0, c0, lane);               // Wt[32][512]
    acc = wmma_f16(a, bf, acc);
  }
  const int h = lane >> 4, n = lane & 15;
  const float bb = bias[c0 + n];
#pragma unroll
  for (int i = 0; i < 8; ++i) {
    size_t r = (size_t)(r0 + h * 8 + i);
    out[r * DP + c0 + n] = acc[i] + bb + pts[r * DP + c0 + n];
  }
}

// ---------------- host-side launch -----------------------------------------

extern "C" void kernel_launch(void* const* d_in, const int* in_sizes, int n_in,
                              void* d_out, int out_size, void* d_ws, size_t ws_size,
                              hipStream_t stream) {
  const float* xyz    = (const float*)d_in[0];
  const float* points = (const float*)d_in[1];
  const float* fc1_w  = (const float*)d_in[2];
  const float* fc1_b  = (const float*)d_in[3];
  const float* fc2_w  = (const float*)d_in[4];
  const float* fc2_b  = (const float*)d_in[5];
  const float* dw1    = (const float*)d_in[6];
  const float* db1    = (const float*)d_in[7];
  const float* dw2    = (const float*)d_in[8];
  const float* db2    = (const float*)d_in[9];
  const float* gw1    = (const float*)d_in[10];
  const float* gb1    = (const float*)d_in[11];
  const float* gw2    = (const float*)d_in[12];
  const float* gb2    = (const float*)d_in[13];
  const float* wq     = (const float*)d_in[14];
  const float* wk     = (const float*)d_in[15];
  const float* wv     = (const float*)d_in[16];
  (void)in_sizes; (void)n_in; (void)out_size; (void)ws_size;

  char* ws = (char*)d_ws;
  size_t off = 0;
  auto walloc = [&](size_t bytes) -> void* {
    void* p = ws + off;
    off = (off + bytes + 255) & ~(size_t)255;
    return p;
  };
  int*       knn    = (int*)      walloc((size_t)RKN * sizeof(int));
  _Float16*  xh     = (_Float16*) walloc((size_t)BN * DM * sizeof(_Float16));
  float*     qbuf   = (float*)    walloc((size_t)BN * DM * sizeof(float));
  float*     kbuf   = (float*)    walloc((size_t)BN * DM * sizeof(float));
  float*     vbuf   = (float*)    walloc((size_t)BN * DM * sizeof(float));
  _Float16*  pos    = (_Float16*) walloc((size_t)RKN * DM * sizeof(_Float16));
  float*     resmid = (float*)    walloc((size_t)BN * DM * sizeof(float));
  _Float16*  fc1_wt = (_Float16*) walloc((size_t)DP * DM * sizeof(_Float16));
  _Float16*  wq_t   = (_Float16*) walloc((size_t)DM * DM * sizeof(_Float16));
  _Float16*  wk_t   = (_Float16*) walloc((size_t)DM * DM * sizeof(_Float16));
  _Float16*  wv_t   = (_Float16*) walloc((size_t)DM * DM * sizeof(_Float16));
  _Float16*  dw2_t  = (_Float16*) walloc((size_t)DM * DM * sizeof(_Float16));
  _Float16*  gw1_t  = (_Float16*) walloc((size_t)DM * DM * sizeof(_Float16));
  _Float16*  gw2_t  = (_Float16*) walloc((size_t)DM * DM * sizeof(_Float16));
  _Float16*  fc2_wt = (_Float16*) walloc((size_t)DM * DP * sizeof(_Float16));

  float* out_res  = (float*)d_out;                 // [BN, 32]
  float* out_attn = out_res + (size_t)BN * DP;     // [RKN, 512]

  // One-time weight transpose + f32->f16 (tiny vs. ~4096x reuse in GEMMs)
  wT_kernel<<<(DP * DM) / 256, 256, 0, stream>>>(fc1_w, fc1_wt, DP, DM);
  wT_kernel<<<(DM * DM) / 256, 256, 0, stream>>>(wq, wq_t, DM, DM);
  wT_kernel<<<(DM * DM) / 256, 256, 0, stream>>>(wk, wk_t, DM, DM);
  wT_kernel<<<(DM * DM) / 256, 256, 0, stream>>>(wv, wv_t, DM, DM);
  wT_kernel<<<(DM * DM) / 256, 256, 0, stream>>>(dw2, dw2_t, DM, DM);
  wT_kernel<<<(DM * DM) / 256, 256, 0, stream>>>(gw1, gw1_t, DM, DM);
  wT_kernel<<<(DM * DM) / 256, 256, 0, stream>>>(gw2, gw2_t, DM, DM);
  wT_kernel<<<(DM * DP) / 256, 256, 0, stream>>>(fc2_w, fc2_wt, DM, DP);

  knn_kernel<<<BN, 32, 0, stream>>>(xyz, knn);
  fc1_kernel<<<1024, 256, 0, stream>>>(points, fc1_wt, fc1_b, xh);
  lin512_kernel<<<256, 256, 0, stream>>>(xh, wq_t, qbuf);
  lin512_kernel<<<256, 256, 0, stream>>>(xh, wk_t, kbuf);
  lin512_kernel<<<256, 256, 0, stream>>>(xh, wv_t, vbuf);
  pos_kernel<<<RKN / 64, 256, 0, stream>>>(xyz, knn, dw1, db1, dw2_t, db2, pos);
  gamma_kernel<<<RKN / 64, 256, 0, stream>>>(qbuf, kbuf, pos, knn, gw1_t, gb1, gw2_t, gb2, out_attn);
  softmax_kernel<<<(BN * DM) / 256, 256, 0, stream>>>(out_attn);
  resmid_kernel<<<(BN * DM) / 256, 256, 0, stream>>>(out_attn, vbuf, pos, knn, resmid);
  fc2_kernel<<<64, 256, 0, stream>>>(resmid, fc2_wt, fc2_b, points, out_res);
}
